// MultiHeadAttention_66468913873512
// MI455X (gfx1250) — matrix operand order repaired and checked by reading, not compiled, not run on hardware
//
#include <hip/hip_runtime.h>
#include <hip/hip_bf16.h>

// Problem constants (match reference)
#define B_  2
#define T_  2048
#define D_  1024
#define H_  16
#define HD_ 64

typedef __attribute__((ext_vector_type(16))) __bf16          v16bf;
typedef __attribute__((ext_vector_type(8)))  float           v8f;
typedef __attribute__((ext_vector_type(16))) unsigned short  v16u;
typedef __attribute__((ext_vector_type(8)))  unsigned short  v8u;
typedef __attribute__((ext_vector_type(4)))  unsigned int    v4u;
typedef __attribute__((ext_vector_type(8)))  int             v8i;
typedef __attribute__((ext_vector_type(4)))  int             v4i;

#if defined(__AMDGCN__) && __has_builtin(__builtin_amdgcn_tensor_load_to_lds)
#define HAVE_TDM 1
#else
#define HAVE_TDM 0
#endif

// f32 -> bf16 bits, round-to-nearest-even
__device__ __forceinline__ unsigned short f2bf(float f) {
  unsigned int u = __builtin_bit_cast(unsigned int, f);
  u += 0x7FFFu + ((u >> 16) & 1u);
  return (unsigned short)(u >> 16);
}

// Low 32 bits of a generic pointer to a __shared__ object == LDS byte offset
// (LDS aperture keeps the offset in addr[31:0] per CDNA5 ISA 10.2).
__device__ __forceinline__ unsigned lds_off(const void* p) {
  return (unsigned)(unsigned long long)p;
}

// One per-lane async copy: 16 bytes global -> LDS, tracked by ASYNCcnt.
__device__ __forceinline__ void async_ld_b128(unsigned short* lds,
                                              const unsigned short* g) {
  unsigned lo = lds_off(lds);
  unsigned long long ga = (unsigned long long)g;
  asm volatile("global_load_async_to_lds_b128 %0, %1, off"
               :: "v"(lo), "v"(ga) : "memory");
}

#if HAVE_TDM
// TDM 2-D tile DMA: tile (tileY rows x tileX elems) of 2-byte elements from a
// row-major tensor with row stride rowStride, into contiguous LDS at ldsOff.
// D# bitfields per CDNA5 ISA 8.3/8.4 (count=1, type=2, data_size=1 -> 2B).
// This toolchain's builtin takes 6 args (g0, g1, g2, g3, g4-pad, cpol).
__device__ __forceinline__ void tdm_load_2d(unsigned ldsOff, const void* gptr,
                                            unsigned tileX, unsigned tileY,
                                            unsigned rowStride) {
  unsigned long long ga = (unsigned long long)gptr;
  v4u g0;
  g0[0] = 1u;                                                  // count=1
  g0[1] = ldsOff;                                              // lds_addr
  g0[2] = (unsigned)ga;                                        // global_addr[31:0]
  g0[3] = (unsigned)((ga >> 32) & 0x01FFFFFFu) | (2u << 30);   // addr[56:32], type=2
  v8i g1;
  g1[0] = (int)(1u << 16);                                     // data_size=1 (2B)
  g1[1] = (int)(tileX << 16);                                  // tensor_dim0 lo16
  g1[2] = (int)((tileX >> 16) | (tileY << 16));                // dim0 hi / dim1 lo
  g1[3] = (int)((tileY >> 16) | (tileX << 16));                // dim1 hi / tile_dim0
  g1[4] = (int)(tileY & 0xFFFFu);                              // tile_dim1 (dim2=0)
  g1[5] = (int)rowStride;                                      // dim0_stride lo32
  g1[6] = 0;
  g1[7] = 0;
  v4i z4 = {0, 0, 0, 0};
  v8i z8 = {0, 0, 0, 0, 0, 0, 0, 0};
  __builtin_amdgcn_tensor_load_to_lds(g0, g1, z4, z4, z8, 0);
}
#endif

// 16x32 bf16 A-fragment from two contiguous 8-elem chunks (A layout:
// lane holds row l%16; j=0..15 -> K=(l/16)*16+j).
__device__ __forceinline__ v16bf frag_a16(const unsigned short* p0,
                                          const unsigned short* p1) {
  v8u lo = *(const v8u*)p0;
  v8u hi = *(const v8u*)p1;
  v16u t;
#pragma unroll
  for (int i = 0; i < 8; ++i) { t[i] = lo[i]; t[i + 8] = hi[i]; }
  return __builtin_bit_cast(v16bf, t);
}

// 32x16 bf16 B-fragment: lane holds column l%16; lanes 0-15 K=0..15,
// lanes 16-31 K=16..31; 16 contiguous elems from K-contiguous staging.
__device__ __forceinline__ v16bf frag_b16(const unsigned short* p) {
  return __builtin_bit_cast(v16bf, *(const v16u*)p);
}

// ---------------------------------------------------------------------------
// Elementwise f32 -> bf16 pre-pack (done once; removes redundant in-GEMM cvt).
// ---------------------------------------------------------------------------
__global__ __launch_bounds__(256) void cvt_bf16_kernel(
    const float* __restrict__ src, unsigned short* __restrict__ dst, int n)
{
  int i = blockIdx.x * 256 + threadIdx.x;
  if (i < n) dst[i] = f2bf(src[i]);
}

// ---------------------------------------------------------------------------
// out[M,N] = A[M,K] @ W[N,K]^T (+ bias[N]); bf16 in (pre-packed), f32 out.
// Block 128x64, 8 waves x (16x64). Double-buffered async global->LDS staging.
// ---------------------------------------------------------------------------
__global__ __launch_bounds__(256) void gemm_bf16_kernel(
    const unsigned short* __restrict__ A, const unsigned short* __restrict__ W,
    const float* __restrict__ bias, float* __restrict__ out,
    int M, int N, int K)
{
  __shared__ __align__(32) unsigned short As[2][128][32];   // [m][k]
  __shared__ __align__(32) unsigned short Bs[2][64][32];    // [n][k]

  const int tid  = threadIdx.x;
  const int wave = tid >> 5;
  const int lane = tid & 31;
  const int lm = lane & 15, kh = lane >> 4;
  const int mBase = blockIdx.y * 128;
  const int nBase = blockIdx.x * 64;

  const int rC = tid >> 2;            // 0..63: chunk row
  const int oC = (tid & 3) << 3;      // 0,8,16,24: chunk col

  auto stage = [&](int buf, int k0) {
    async_ld_b128(&As[buf][rC][oC],      A + (size_t)(mBase + rC) * K + k0 + oC);
    async_ld_b128(&As[buf][64 + rC][oC], A + (size_t)(mBase + 64 + rC) * K + k0 + oC);
    async_ld_b128(&Bs[buf][rC][oC],      W + (size_t)(nBase + rC) * K + k0 + oC);
  };

  v8f acc[4] = {};
  stage(0, 0);
  for (int k0 = 0; k0 < K; k0 += 32) {
    const int buf = (k0 >> 5) & 1;
    const bool more = (k0 + 32) < K;
    if (more) {
      stage(buf ^ 1, k0 + 32);
      asm volatile("s_wait_asynccnt 0x3" ::: "memory");  // current panel done
    } else {
      asm volatile("s_wait_asynccnt 0x0" ::: "memory");
    }
    __syncthreads();

    v16bf af = frag_a16(&As[buf][wave * 16 + lm][kh * 16],
                        &As[buf][wave * 16 + lm][kh * 16 + 8]);
#pragma unroll
    for (int nt = 0; nt < 4; ++nt) {
      v16bf bf = frag_b16(&Bs[buf][nt * 16 + lm][kh * 16]);
      acc[nt] = __builtin_amdgcn_wmma_f32_16x16x32_bf16(
          false, af, false, bf, (short)0, acc[nt], false, false);
    }
    __syncthreads();   // protect buffer for next iteration's async issue
  }

  const int rowHalf = kh * 8;
#pragma unroll
  for (int nt = 0; nt < 4; ++nt) {
    int n = nBase + nt * 16 + lm;
    float bvv = bias ? bias[n] : 0.0f;
#pragma unroll
    for (int r = 0; r < 8; ++r) {
      int mrow = mBase + wave * 16 + rowHalf + r;
      out[(size_t)mrow * N + n] = acc[nt][r] + bvv;
    }
  }
}

// ---------------------------------------------------------------------------
// RoPE + scale on Q/K, repack Q/K/V to bf16 (B,H,T,HD).
// ---------------------------------------------------------------------------
__global__ __launch_bounds__(256) void rope_pack_kernel(
    const float* __restrict__ qf, const float* __restrict__ kf,
    const float* __restrict__ vf,
    const float* __restrict__ cosT, const float* __restrict__ sinT,
    unsigned short* __restrict__ qb, unsigned short* __restrict__ kb,
    unsigned short* __restrict__ vb)
{
  const int HD2 = HD_ / 2;
  size_t idx = (size_t)blockIdx.x * blockDim.x + threadIdx.x;
  int i = (int)(idx % HD2);
  int h = (int)((idx / HD2) % H_);
  int t = (int)((idx / ((size_t)HD2 * H_)) % T_);
  int b = (int)(idx / ((size_t)HD2 * H_ * T_));
  const float scale = 0.3535533905932738f;  // 64^-0.25

  size_t src = (((size_t)b * T_ + t) * H_ + h) * HD_ + 2 * i;
  float c = cosT[(size_t)t * HD2 + i];
  float s = sinT[(size_t)t * HD2 + i];
  float q0 = qf[src], q1 = qf[src + 1];
  float k0 = kf[src], k1 = kf[src + 1];

  size_t dst = (((size_t)b * H_ + h) * T_ + t) * HD_ + 2 * i;
  qb[dst]     = f2bf((q0 * c - q1 * s) * scale);
  qb[dst + 1] = f2bf((q0 * s + q1 * c) * scale);
  kb[dst]     = f2bf((k0 * c - k1 * s) * scale);
  kb[dst + 1] = f2bf((k0 * s + k1 * c) * scale);
  vb[dst]     = f2bf(vf[src]);
  vb[dst + 1] = f2bf(vf[src + 1]);
}

// ---------------------------------------------------------------------------
// Flash-style attention per (b,h,128-row block). K/V blocks staged by the
// Tensor Data Mover (fallback: per-lane async loads), double-buffered.
// Writes raw masked logits to qk; O = softmax(S)@V written as bf16 (B,T,D).
// ---------------------------------------------------------------------------
__global__ __launch_bounds__(256) void attn_kernel(
    const unsigned short* __restrict__ qb, const unsigned short* __restrict__ kb,
    const unsigned short* __restrict__ vb,
    float* __restrict__ qk, unsigned short* __restrict__ wv)
{
  __shared__ __align__(32) unsigned short kBuf[2][32][64];   // [s][d]
  __shared__ __align__(32) unsigned short vBuf[2][32][64];   // [s][d]
  __shared__ __align__(32) unsigned short pS[8][16][32];     // per-wave P tile

  const int tid  = threadIdx.x;
  const int wave = tid >> 5;
  const int lane = tid & 31;
  const int lm = lane & 15, kh = lane >> 4, rowHalf = kh * 8;

  const int nTB  = T_ / 128;
  const int tBlk = blockIdx.x % nTB;
  const int h    = (blockIdx.x / nTB) % H_;
  const int b    = blockIdx.x / (nTB * H_);
  const size_t bh = (size_t)(b * H_ + h);

  const unsigned short* qBase = qb + bh * T_ * HD_;
  const unsigned short* kBase = kb + bh * T_ * HD_;
  const unsigned short* vBase = vb + bh * T_ * HD_;
  float* qkBase = qk + bh * (size_t)T_ * T_;
  const int tRow = tBlk * 128 + wave * 16;

  auto stage = [&](int buf, int s0) {
#if HAVE_TDM
    if (wave == 0)
      tdm_load_2d(lds_off(&kBuf[buf][0][0]), kBase + (size_t)s0 * HD_,
                  HD_, 32, HD_);
    else if (wave == 1)
      tdm_load_2d(lds_off(&vBuf[buf][0][0]), vBase + (size_t)s0 * HD_,
                  HD_, 32, HD_);
#else
    int r = tid >> 3, co = (tid & 7) << 3;
    async_ld_b128(&kBuf[buf][r][co], kBase + (size_t)(s0 + r) * HD_ + co);
    async_ld_b128(&vBuf[buf][r][co], vBase + (size_t)(s0 + r) * HD_ + co);
#endif
  };

  // Q fragments for this wave's 16 rows, K(=d) split 0..31 / 32..63.
  v16bf qa[2];
#pragma unroll
  for (int df = 0; df < 2; ++df) {
    const unsigned short* p = qBase + (size_t)(tRow + lm) * HD_ + df * 32;
    qa[df] = frag_a16(p + kh * 16, p + kh * 16 + 8);
  }

  float mrow[8], lrow[8];
  v8f o[4] = {};
#pragma unroll
  for (int r = 0; r < 8; ++r) { mrow[r] = -1e30f; lrow[r] = 0.0f; }

  stage(0, 0);
  for (int s0 = 0; s0 < T_; s0 += 32) {
    const int buf = (s0 >> 5) & 1;
    const bool more = (s0 + 32) < T_;
    if (more) stage(buf ^ 1, s0 + 32);
#if HAVE_TDM
    if (more) __builtin_amdgcn_s_wait_tensorcnt((short)1);
    else      __builtin_amdgcn_s_wait_tensorcnt((short)0);
#else
    if (more) asm volatile("s_wait_asynccnt 0x2" ::: "memory");
    else      asm volatile("s_wait_asynccnt 0x0" ::: "memory");
#endif
    __syncthreads();

    // S = Q @ K^T: two 16x16 column tiles, K-dim = HD = 64 -> 2 WMMAs each.
    v8f sC[2];
#pragma unroll
    for (int st = 0; st < 2; ++st) {
      v8f c = {};
#pragma unroll
      for (int df = 0; df < 2; ++df) {
        v16bf bf = frag_b16(&kBuf[buf][st * 16 + lm][df * 32 + kh * 16]);
        c = __builtin_amdgcn_wmma_f32_16x16x32_bf16(
            false, qa[df], false, bf, (short)0, c, false, false);
      }
      sC[st] = c;
    }

    // Causal mask, stream raw logits to qk, per-row tile max.
    float tmax[8];
#pragma unroll
    for (int r = 0; r < 8; ++r) {
      int t = tRow + rowHalf + r;
      int sa = s0 + lm, sb = s0 + 16 + lm;
      float va  = sC[0][r] + ((sa > t) ? -1000000000.0f : 0.0f);
      float vbv = sC[1][r] + ((sb > t) ? -1000000000.0f : 0.0f);
      qkBase[(size_t)t * T_ + sa] = va;
      qkBase[(size_t)t * T_ + sb] = vbv;
      sC[0][r] = va; sC[1][r] = vbv;
      tmax[r] = fmaxf(va, vbv);
    }
    // Reduce over 16 lanes of each half (wave32: xor masks < 16 stay in-half).
#pragma unroll
    for (int msk = 1; msk < 16; msk <<= 1)
#pragma unroll
      for (int r = 0; r < 8; ++r)
        tmax[r] = fmaxf(tmax[r], __shfl_xor(tmax[r], msk, 32));

    float alpha[8], psum[8];
#pragma unroll
    for (int r = 0; r < 8; ++r) {
      float mn = fmaxf(mrow[r], tmax[r]);
      alpha[r] = __expf(mrow[r] - mn);
      mrow[r] = mn;
      float p0 = __expf(sC[0][r] - mn);
      float p1 = __expf(sC[1][r] - mn);
      pS[wave][rowHalf + r][lm]      = f2bf(p0);
      pS[wave][rowHalf + r][16 + lm] = f2bf(p1);
      psum[r] = p0 + p1;
    }
#pragma unroll
    for (int msk = 1; msk < 16; msk <<= 1)
#pragma unroll
      for (int r = 0; r < 8; ++r)
        psum[r] += __shfl_xor(psum[r], msk, 32);
#pragma unroll
    for (int r = 0; r < 8; ++r) lrow[r] = lrow[r] * alpha[r] + psum[r];
#pragma unroll
    for (int nt = 0; nt < 4; ++nt)
#pragma unroll
      for (int r = 0; r < 8; ++r)
        o[nt][r] *= alpha[r];

    // O += P(16x32) @ V(32x64). P restriped via LDS (same-wave DS in-order).
    v16bf pa = frag_a16(&pS[wave][lm][kh * 16], &pS[wave][lm][kh * 16 + 8]);
#pragma unroll
    for (int nt = 0; nt < 4; ++nt) {
      v16u tv;
#pragma unroll
      for (int j = 0; j < 16; ++j) tv[j] = vBuf[buf][kh * 16 + j][nt * 16 + lm];
      v16bf bfv = __builtin_bit_cast(v16bf, tv);
      o[nt] = __builtin_amdgcn_wmma_f32_16x16x32_bf16(
          false, pa, false, bfv, (short)0, o[nt], false, false);
    }
    __syncthreads();   // protect buffers for next iteration's stage()
  }

  // Normalize; write wv directly as bf16 (B,T,H,HD) == (B,T,D).
#pragma unroll
  for (int nt = 0; nt < 4; ++nt)
#pragma unroll
    for (int r = 0; r < 8; ++r) {
      int t = tRow + rowHalf + r;
      int d = nt * 16 + lm;
      wv[(((size_t)b * T_ + t) * H_ + h) * HD_ + d] = f2bf(o[nt][r] / lrow[r]);
    }
}

// ---------------------------------------------------------------------------
// Launch. Inputs: x, mask(unused; analytic), cos, sin, Wq, bq, Wk, Wv, bv,
// Wo, bo.  d_out = [out (B*T*D f32) | qk (B*H*T*T f32)].  ws: 96 MiB.
// ---------------------------------------------------------------------------
extern "C" void kernel_launch(void* const* d_in, const int* in_sizes, int n_in,
                              void* d_out, int out_size, void* d_ws, size_t ws_size,
                              hipStream_t stream)
{
  (void)in_sizes; (void)n_in; (void)out_size; (void)ws_size;

  const float* x    = (const float*)d_in[0];
  const float* cosT = (const float*)d_in[2];
  const float* sinT = (const float*)d_in[3];
  const float* Wq   = (const float*)d_in[4];
  const float* bq   = (const float*)d_in[5];
  const float* Wk   = (const float*)d_in[6];
  const float* Wv   = (const float*)d_in[7];
  const float* bv   = (const float*)d_in[8];
  const float* Wo   = (const float*)d_in[9];
  const float* bo   = (const float*)d_in[10];

  float* out = (float*)d_out;
  float* qk  = out + (size_t)B_ * T_ * D_;

  const size_t fElems = (size_t)B_ * T_ * D_;   // 4.19M
  const size_t wElems = (size_t)D_ * D_;        // 1.05M
  float* qf  = (float*)d_ws;
  float* kf  = qf + fElems;
  float* vf  = kf + fElems;
  unsigned short* qbuf = (unsigned short*)(vf + fElems);
  unsigned short* kbuf = qbuf + fElems;
  unsigned short* vbuf = kbuf + fElems;
  unsigned short* wvb  = vbuf + fElems;
  unsigned short* xb   = wvb + fElems;
  unsigned short* Wqb  = xb + fElems;
  unsigned short* Wkb  = Wqb + wElems;
  unsigned short* Wvb  = Wkb + wElems;
  unsigned short* Wob  = Wvb + wElems;

  // Pre-pack operands to bf16 (once).
  cvt_bf16_kernel<<<(int)(fElems / 256), 256, 0, stream>>>(x,  xb,  (int)fElems);
  cvt_bf16_kernel<<<(int)(wElems / 256), 256, 0, stream>>>(Wq, Wqb, (int)wElems);
  cvt_bf16_kernel<<<(int)(wElems / 256), 256, 0, stream>>>(Wk, Wkb, (int)wElems);
  cvt_bf16_kernel<<<(int)(wElems / 256), 256, 0, stream>>>(Wv, Wvb, (int)wElems);
  cvt_bf16_kernel<<<(int)(wElems / 256), 256, 0, stream>>>(Wo, Wob, (int)wElems);

  const int M = B_ * T_, N = D_, K = D_;
  dim3 gg(N / 64, M / 128);
  gemm_bf16_kernel<<<gg, 256, 0, stream>>>(xb, Wqb, bq,      qf, M, N, K);
  gemm_bf16_kernel<<<gg, 256, 0, stream>>>(xb, Wkb, nullptr, kf, M, N, K);
  gemm_bf16_kernel<<<gg, 256, 0, stream>>>(xb, Wvb, bv,      vf, M, N, K);

  size_t nRope = (size_t)B_ * T_ * H_ * (HD_ / 2);
  rope_pack_kernel<<<(int)((nRope + 255) / 256), 256, 0, stream>>>(
      qf, kf, vf, cosT, sinT, qbuf, kbuf, vbuf);

  attn_kernel<<<B_ * H_ * (T_ / 128), 256, 0, stream>>>(qbuf, kbuf, vbuf, qk, wvb);

  // Final projection consumes bf16 wv directly.
  gemm_bf16_kernel<<<gg, 256, 0, stream>>>(wvb, Wob, bo, out, M, N, K);
}
